// SocialItemGraph_1821066134230
// MI455X (gfx1250) — compile-verified, matrix operand order checked by reading, not run on hardware
//
#include <hip/hip_runtime.h>
#include <hip/hip_bf16.h>
#include <math.h>

#define N_ITEMS 7000
#define N_NODES 37000
#define DIM     128
#define NEDGE   400000
#define NREL    10
#define NBASE   8
#define BATCH   512
#define CTXL    100

typedef __attribute__((ext_vector_type(16))) _Float16     v16h;
typedef __attribute__((ext_vector_type(8)))  float        v8f;
typedef __attribute__((ext_vector_type(4)))  unsigned int u32x4;
typedef __attribute__((ext_vector_type(8)))  unsigned int u32x8;

// K index inside a 32-wide K-chunk for element j of a 16-bit A/B WMMA fragment
// (wave32 layout, CDNA5 ISA 7.12.2): lanes 0-15 hold K 0-7,16-23;
// lanes 16-31 hold K 8-15,24-31.
__device__ __forceinline__ int kmap16(int j, int hi) {
  return (j < 8 ? j : j + 8) + (hi ? 8 : 0);
}

// convert float4 -> 4 halfs, scaled by mask (branch-free invalid-row zeroing)
__device__ __forceinline__ void cvt4m(v16h& a, int base, float4 f, float m) {
  a[base + 0] = (_Float16)(f.x * m);
  a[base + 1] = (_Float16)(f.y * m);
  a[base + 2] = (_Float16)(f.z * m);
  a[base + 3] = (_Float16)(f.w * m);
}

// CDNA5 has a V_TANH_F32 transcendental; use it when the builtin exists.
__device__ __forceinline__ float fast_tanh(float x) {
#if __has_builtin(__builtin_amdgcn_tanhf)
  return __builtin_amdgcn_tanhf(x);
#else
  float e = __expf(-2.0f * x);          // v_exp_f32 path, no libm branches
  return (1.0f - e) / (1.0f + e);
#endif
}

// 1-D Tensor Data Mover copy: nbytes (multiple of 8) from global `gptr` into
// LDS byte-offset `ldsaddr`. D# per CDNA5 ISA:
//   group0: count=1 | lds_addr | global_addr[56:0] | type=2
//   group1: data_size=3 (8B), tensor_dim0=tile_dim0=nbytes/8, dim1=1,
//           tensor_dim0_stride=nbytes/8
__device__ __forceinline__ void tdm_load_1d(unsigned ldsaddr, const void* gptr,
                                            unsigned nbytes) {
  unsigned nel = nbytes >> 3;
  unsigned long long ga = (unsigned long long)(uintptr_t)gptr;
  u32x4 g0;
  g0[0] = 1u;                                   // count=1, user descriptor
  g0[1] = ldsaddr;                              // lds_addr
  g0[2] = (unsigned)(ga & 0xFFFFFFFFull);       // global_addr[31:0]
  g0[3] = (unsigned)((ga >> 32) & 0x01FFFFFFull) | (2u << 30);  // [56:32]|type=2
  u32x8 g1;
  g1[0] = 3u << 16;                             // data_size = 8B
  g1[1] = (nel & 0xFFFFu) << 16;                // tensor_dim0[15:0]
  g1[2] = ((nel >> 16) & 0xFFFFu) | (1u << 16); // tensor_dim0[31:16], dim1=1
  g1[3] = (nel & 0xFFFFu) << 16;                // tile_dim0
  g1[4] = 0u;                                   // tile_dim1/2 unused
  g1[5] = nel;                                  // tensor_dim0_stride[31:0]
  g1[6] = 0u;
  g1[7] = 0u;
  asm volatile("tensor_load_to_lds %0, %1" ::"s"(g0), "s"(g1) : "memory");
}

// ---------------------------------------------------------------- utilities
__global__ void k_zero(float* p, long long n) {
  long long i = (long long)blockIdx.x * blockDim.x + threadIdx.x;
  if (i < n) p[i] = 0.0f;
}

// Convert f32 row-major [K,N] weights into WMMA-B fragment order:
// half index = ((nt*KC + kc)*32 + lane)*16 + j  ->  W[kc*32+kmap, nt*16+lm]
__global__ void k_swizzle_w(const float* W, _Float16* Wsw, int K, int N) {
  int i = blockIdx.x * blockDim.x + threadIdx.x;
  if (i >= K * N) return;
  int j    = i & 15;
  int lane = (i >> 4) & 31;
  int fc   = i >> 9;
  int KC   = K >> 5;
  int nt   = fc / KC;
  int kc   = fc - nt * KC;
  int k    = kc * 32 + kmap16(j, lane >> 4);
  int n    = nt * 16 + (lane & 15);
  Wsw[i] = (_Float16)W[k * N + n];
}

// ------------------------------------------------------------------- RGCN
__global__ void k_count(const int* ei, const int* et, float* cnt, float* icnt) {
  int e = blockIdx.x * blockDim.x + threadIdx.x;
  if (e >= NEDGE) return;
  int src = ei[e];
  int dst = ei[NEDGE + e];
  int r   = et[e];
  atomicAdd(&cnt[dst * NREL + r], 1.0f);
  atomicAdd(&icnt[src], 1.0f);
}

__global__ __launch_bounds__(128)
void k_msg(const int* ei, const int* et, const float* basis, const float* comp,
           const float* cnt, float* agg) {
  int e = blockIdx.x;
  int d = threadIdx.x;
  int src = ei[e], dst = ei[NEDGE + e], r = et[e];
  __builtin_prefetch(agg + (size_t)dst * DIM + d, 0, 3);  // warm scatter target
  float inv = 1.0f / fmaxf(cnt[dst * NREL + r], 1.0f);
  float m = 0.0f;
#pragma unroll
  for (int b = 0; b < NBASE; ++b)
    m += comp[r * NBASE + b] * basis[((size_t)b * N_NODES + src) * DIM + d];
  atomicAdd(&agg[(size_t)dst * DIM + d], m * inv);
}

__global__ void k_node_emb(float* agg, const float* root, const float* bias,
                           long long n) {
  long long i = (long long)blockIdx.x * blockDim.x + threadIdx.x;
  if (i < n) agg[i] += root[i] + bias[i & (DIM - 1)];
}

__global__ __launch_bounds__(128)
void k_isum(const int* ei, const float* node_emb, float* isum) {
  int e = blockIdx.x, d = threadIdx.x;
  int src = ei[e], dst = ei[NEDGE + e];
  atomicAdd(&isum[(size_t)src * DIM + d], node_emb[(size_t)dst * DIM + d]);
}

__global__ void k_item_mean(float* isum, const float* icnt, int n) {
  int i = blockIdx.x * blockDim.x + threadIdx.x;
  if (i < n) isum[i] /= fmaxf(icnt[i >> 7], 1.0f);
}

// ------------------------------------------------ attention scores (WMMA)
// e[row] = sum_n tanh((h @ A)[row,n]) * attn_b[n]; 8 waves/block, A staged in
// LDS via TDM, B-fragments software-pipelined one ahead.
__global__ __launch_bounds__(256)
void k_score(const int* ctx, const float* item_mean, const float* icnt,
             const _Float16* Asw, const float* attn_b, float* e_out) {
  extern __shared__ _Float16 wlds[];
  const int tid = threadIdx.x;
  if ((tid >> 5) == 0) {  // wave 0 issues the tensor DMA (EXEC ignored)
    tdm_load_1d((unsigned)(uintptr_t)wlds, Asw, DIM * DIM * 2);
    __builtin_amdgcn_s_wait_tensorcnt(0);
  }
  __syncthreads();

  const int lane = tid & 31;
  const int wv   = tid >> 5;
  const int hi   = lane >> 4;
  const int lm   = lane & 15;
  const long long tile0 = (long long)blockIdx.x * 128 + wv * 16;

  long long row = tile0 + lm;
  int it = ctx[row];
  bool valid = (it >= 0);
  int ic = valid ? it : 0;
  valid = valid && (icnt[ic] > 0.0f);
  float msk = valid ? 1.0f : 0.0f;              // branch-free masking
  const float* hrow = item_mean + (size_t)ic * DIM;

  v16h a[4];
#pragma unroll
  for (int kc = 0; kc < 4; ++kc) {
    const float4* q0 = (const float4*)(hrow + kc * 32 + hi * 8);
    const float4* q1 = (const float4*)(hrow + kc * 32 + 16 + hi * 8);
    cvt4m(a[kc], 0,  q0[0], msk);
    cvt4m(a[kc], 4,  q0[1], msk);
    cvt4m(a[kc], 8,  q1[0], msk);
    cvt4m(a[kc], 12, q1[1], msk);
  }

  float esum[8];
#pragma unroll
  for (int v = 0; v < 8; ++v) esum[v] = 0.0f;

  const v16h* wfr = (const v16h*)wlds;
  v16h bf = wfr[lane];                          // prefetch fragment 0
#pragma unroll
  for (int nt = 0; nt < DIM / 16; ++nt) {
    v8f acc = {};
#pragma unroll
    for (int kc = 0; kc < 4; ++kc) {
      v16h cur = bf;
      int nxt = nt * 4 + kc + 1;
      if (nxt < 32) bf = wfr[nxt * 32 + lane];  // overlap ds load with WMMA
      acc = __builtin_amdgcn_wmma_f32_16x16x32_f16(
          false, a[kc], false, cur, (short)0, acc, false, false);
    }
    float bn = attn_b[nt * 16 + lm];
#pragma unroll
    for (int v = 0; v < 8; ++v) esum[v] += fast_tanh(acc[v]) * bn;
  }
#pragma unroll
  for (int v = 0; v < 8; ++v) {
    float s = esum[v];
    s += __shfl_xor(s, 1, 32);
    s += __shfl_xor(s, 2, 32);
    s += __shfl_xor(s, 4, 32);
    s += __shfl_xor(s, 8, 32);
    if (lm == 0) e_out[tile0 + v + 8 * hi] = s;
  }
}

// ----------------------------------------- masked softmax + weighted pool
__global__ __launch_bounds__(128)
void k_attn_pool(const int* ctx, const float* e, const float* item_mean,
                 const float* icnt, float* pooled) {
  __shared__ float w[CTXL];
  __shared__ int   idx[CTXL];
  int b = blockIdx.x, t = threadIdx.x;
  if (t < CTXL) {
    int it = ctx[b * CTXL + t];
    bool valid = (it >= 0);
    int ic = valid ? it : 0;
    valid = valid && (icnt[ic] > 0.0f);
    idx[t] = valid ? ic : -1;
    w[t]   = valid ? e[b * CTXL + t] : -1e9f;
  }
  __syncthreads();
  if (t == 0) {
    float mx = -1e30f;
    for (int l = 0; l < CTXL; ++l) mx = fmaxf(mx, w[l]);
    float s = 0.0f;
    for (int l = 0; l < CTXL; ++l) { float ex = __expf(w[l] - mx); w[l] = ex; s += ex; }
    float invs = 1.0f / s;
    for (int l = 0; l < CTXL; ++l) w[l] = (idx[l] >= 0) ? w[l] * invs : 0.0f;
  }
  __syncthreads();
  float acc = 0.0f;
  for (int l = 0; l < CTXL; ++l) {
    float wl = w[l];
    if (wl != 0.0f) acc += wl * item_mean[(size_t)idx[l] * DIM + t];
  }
  pooled[(size_t)b * DIM + t] = acc;
}

// ---------------------------------------- fused GEMM+bias+ReLU (WMMA)
// Y[M,128] = relu(X[M,K] @ W + bias); W pre-swizzled, TDM-staged in LDS.
// 4 waves/block, 16 rows/wave; B-fragments software-pipelined.
template <int K>
__global__ __launch_bounds__(128)
void k_gemm(const float* X, const _Float16* Wsw, const float* bias, float* Y,
            int do_relu) {
  constexpr int N     = DIM;
  constexpr int KC    = K / 32;
  constexpr int NFRAG = (N / 16) * KC;
  extern __shared__ _Float16 wlds[];
  const int tid = threadIdx.x;
  if ((tid >> 5) == 0) {
    tdm_load_1d((unsigned)(uintptr_t)wlds, Wsw, K * N * 2);
    __builtin_amdgcn_s_wait_tensorcnt(0);
  }
  __syncthreads();

  const int lane = tid & 31;
  const int wv   = tid >> 5;
  const int hi   = lane >> 4;
  const int lm   = lane & 15;
  const long long tile0 = (long long)blockIdx.x * 64 + wv * 16;

  const float* xrow = X + (tile0 + lm) * K;
  v16h a[KC];
#pragma unroll
  for (int kc = 0; kc < KC; ++kc) {
    const float4* q0 = (const float4*)(xrow + kc * 32 + hi * 8);
    const float4* q1 = (const float4*)(xrow + kc * 32 + 16 + hi * 8);
    cvt4m(a[kc], 0,  q0[0], 1.0f);
    cvt4m(a[kc], 4,  q0[1], 1.0f);
    cvt4m(a[kc], 8,  q1[0], 1.0f);
    cvt4m(a[kc], 12, q1[1], 1.0f);
  }

  const v16h* wfr = (const v16h*)wlds;
  v16h bf = wfr[lane];
#pragma unroll
  for (int nt = 0; nt < N / 16; ++nt) {
    v8f acc = {};
#pragma unroll
    for (int kc = 0; kc < KC; ++kc) {
      v16h cur = bf;
      int nxt = nt * KC + kc + 1;
      if (nxt < NFRAG) bf = wfr[nxt * 32 + lane];
      acc = __builtin_amdgcn_wmma_f32_16x16x32_f16(
          false, a[kc], false, cur, (short)0, acc, false, false);
    }
    int n = nt * 16 + lm;
    float bb = bias[n];
#pragma unroll
    for (int v = 0; v < 8; ++v) {
      float y = acc[v] + bb;
      if (do_relu) y = fmaxf(y, 0.0f);
      Y[(tile0 + v + 8 * hi) * N + n] = y;
    }
  }
}

__global__ void k_concat(const float* p0, const float* p1, float* both) {
  int i = blockIdx.x * blockDim.x + threadIdx.x;
  if (i >= BATCH * DIM) return;
  int b = i >> 7, d = i & (DIM - 1);
  both[b * 2 * DIM + d] = p0[i];
  both[b * 2 * DIM + DIM + d] = p1[i];
}

// --------------------------------------------------------------- dispatch
extern "C" void kernel_launch(void* const* d_in, const int* in_sizes, int n_in,
                              void* d_out, int out_size, void* d_ws,
                              size_t ws_size, hipStream_t stream) {
  const int*   edge_index = (const int*)d_in[0];
  const int*   edge_type  = (const int*)d_in[1];
  const int*   ctx_init   = (const int*)d_in[2];
  const int*   ctx_resp   = (const int*)d_in[3];
  const float* basis  = (const float*)d_in[4];
  const float* comp   = (const float*)d_in[5];
  const float* root   = (const float*)d_in[6];
  const float* bias   = (const float*)d_in[7];
  const float* attn_a = (const float*)d_in[8];
  const float* attn_b = (const float*)d_in[9];
  const float* fc1_w  = (const float*)d_in[10];
  const float* fc1_b  = (const float*)d_in[11];
  const float* fc2_w  = (const float*)d_in[12];
  const float* fc2_b  = (const float*)d_in[13];
  const float* efc1_w = (const float*)d_in[14];
  const float* efc1_b = (const float*)d_in[15];
  const float* efc2_w = (const float*)d_in[16];
  const float* efc2_b = (const float*)d_in[17];
  float* out = (float*)d_out;

  float* ws = (float*)d_ws;
  float* cnt  = ws; ws += N_NODES * NREL;
  float* icnt = ws; ws += N_ITEMS;
  float* agg  = ws; ws += (size_t)N_NODES * DIM;   // becomes node_emb
  float* isum = ws; ws += (size_t)N_ITEMS * DIM;   // becomes item_mean
  long long zn = (long long)N_NODES * NREL + N_ITEMS +
                 (long long)N_NODES * DIM + (long long)N_ITEMS * DIM;
  float* e0 = ws; ws += BATCH * CTXL;
  float* e1 = ws; ws += BATCH * CTXL;
  float* pooled0 = ws; ws += BATCH * DIM;
  float* pooled1 = ws; ws += BATCH * DIM;
  float* hid  = ws; ws += BATCH * DIM;
  float* both = ws; ws += BATCH * 2 * DIM;
  _Float16* h16 = (_Float16*)ws;
  _Float16* attnA16 = h16; h16 += DIM * DIM;
  _Float16* fc1_16  = h16; h16 += DIM * DIM;
  _Float16* fc2_16  = h16; h16 += DIM * DIM;
  _Float16* efc1_16 = h16; h16 += 2 * DIM * DIM;
  _Float16* efc2_16 = h16; h16 += DIM * DIM;

  k_zero<<<(int)((zn + 255) / 256), 256, 0, stream>>>(cnt, zn);
  k_swizzle_w<<<(DIM * DIM + 255) / 256, 256, 0, stream>>>(attn_a, attnA16, DIM, DIM);
  k_swizzle_w<<<(DIM * DIM + 255) / 256, 256, 0, stream>>>(fc1_w, fc1_16, DIM, DIM);
  k_swizzle_w<<<(DIM * DIM + 255) / 256, 256, 0, stream>>>(fc2_w, fc2_16, DIM, DIM);
  k_swizzle_w<<<(2 * DIM * DIM + 255) / 256, 256, 0, stream>>>(efc1_w, efc1_16, 2 * DIM, DIM);
  k_swizzle_w<<<(DIM * DIM + 255) / 256, 256, 0, stream>>>(efc2_w, efc2_16, DIM, DIM);

  // RGCN
  k_count<<<(NEDGE + 255) / 256, 256, 0, stream>>>(edge_index, edge_type, cnt, icnt);
  k_msg<<<NEDGE, 128, 0, stream>>>(edge_index, edge_type, basis, comp, cnt, agg);
  long long nn = (long long)N_NODES * DIM;
  k_node_emb<<<(int)((nn + 255) / 256), 256, 0, stream>>>(agg, root, bias, nn);
  k_isum<<<NEDGE, 128, 0, stream>>>(edge_index, agg, isum);
  k_item_mean<<<(N_ITEMS * DIM + 255) / 256, 256, 0, stream>>>(isum, icnt, N_ITEMS * DIM);

  // attention (WMMA, TDM-staged weights) + pooling
  const int score_lds = DIM * DIM * 2;
  k_score<<<BATCH * CTXL / 128, 256, score_lds, stream>>>(ctx_init, isum, icnt, attnA16, attn_b, e0);
  k_score<<<BATCH * CTXL / 128, 256, score_lds, stream>>>(ctx_resp, isum, icnt, attnA16, attn_b, e1);
  k_attn_pool<<<BATCH, 128, 0, stream>>>(ctx_init, e0, isum, icnt, pooled0);
  k_attn_pool<<<BATCH, 128, 0, stream>>>(ctx_resp, e1, isum, icnt, pooled1);

  // projections (WMMA GEMMs). d_out = [profile | p_init | p_resp]
  float* p_init = out + BATCH * DIM;
  float* p_resp = out + 2 * BATCH * DIM;
  const int lds1 = DIM * DIM * 2, lds2 = 2 * DIM * DIM * 2;
  k_gemm<DIM><<<BATCH / 64, 128, lds1, stream>>>(pooled0, fc1_16, fc1_b, hid, 1);
  k_gemm<DIM><<<BATCH / 64, 128, lds1, stream>>>(hid, fc2_16, fc2_b, p_init, 1);
  k_gemm<DIM><<<BATCH / 64, 128, lds1, stream>>>(pooled1, fc1_16, fc1_b, hid, 1);
  k_gemm<DIM><<<BATCH / 64, 128, lds1, stream>>>(hid, fc2_16, fc2_b, p_resp, 1);
  k_concat<<<(BATCH * DIM + 255) / 256, 256, 0, stream>>>(p_init, p_resp, both);
  k_gemm<2 * DIM><<<BATCH / 64, 128, lds2, stream>>>(both, efc1_16, efc1_b, hid, 1);
  k_gemm<DIM><<<BATCH / 64, 128, lds1, stream>>>(hid, efc2_16, efc2_b, out, 1);
}
